// AudioXMMDiTCrossAttention_61881888801241
// MI455X (gfx1250) — matrix operand order repaired and checked by reading, not compiled, not run on hardware
//
#include <hip/hip_runtime.h>
#include <hip/hip_bf16.h>
#include <stdint.h>

// ---------------------------------------------------------------------------
// AudioX MMDiT cross-attention, CDNA5 (gfx1250), bf16 WMMA pipeline.
//   dim=2048, heads=32, head_dim=64, b=2, n=4096, m=512
// Pipeline: f32->bf16 converts; weight permutation folded into Wk/Wv prep;
// WMMA GEMMs with fused per-head RMSNorm; flash attention with online softmax.
// All matrix math on v_wmma_f32_16x16x32_bf16; k-loops software-pipelined so
// WMMAs overlap with the next chunk's global loads.
// ---------------------------------------------------------------------------

#define DIM 2048
#define NH  32
#define HD  64
#define NQ  4096
#define NM  512
#define NB  2

typedef __attribute__((ext_vector_type(16))) __bf16        v16bf;
typedef __attribute__((ext_vector_type(8)))  float         v8f;
typedef __attribute__((ext_vector_type(8)))  unsigned int  v8u;

static __device__ __forceinline__ unsigned short f2bf(float f) {
  unsigned int u = __float_as_uint(f);
  u += 0x7FFFu + ((u >> 16) & 1u);            // round-to-nearest-even
  return (unsigned short)(u >> 16);
}

static __device__ __forceinline__ v16bf pack_frag(uint4 lo, uint4 hi) {
  v8u r;
  r[0] = lo.x; r[1] = lo.y; r[2] = lo.z; r[3] = lo.w;
  r[4] = hi.x; r[5] = hi.y; r[6] = hi.z; r[7] = hi.w;
  return __builtin_bit_cast(v16bf, r);
}

// A-matrix fragment 16x32 bf16 (M x K). p points at (row = lane&15)'s K-chunk
// base. Lanes 0-15 hold K {0..7, 16..23}; lanes 16-31 hold K {8..15, 24..31}.
static __device__ __forceinline__ v16bf load_fragA(const unsigned short* p, int lane) {
  const int klo = (lane & 16) ? 8 : 0;
  uint4 lo = *(const uint4*)(p + klo);
  uint4 hi = *(const uint4*)(p + klo + 16);
  return pack_frag(lo, hi);
}

// B-matrix fragment 32x16 bf16 (K x N). p points at (col = lane&15)'s K run
// (contiguous in memory). Lanes 0-15 hold K {0..7, 16..23}; lanes 16-31 hold K {8..15, 24..31}.
static __device__ __forceinline__ v16bf load_fragB(const unsigned short* p, int lane) {
  const int koff = (lane & 16) ? 8 : 0;
  uint4 lo = *(const uint4*)(p + koff);
  uint4 hi = *(const uint4*)(p + koff + 16);
  return pack_frag(lo, hi);
}

static __device__ __forceinline__ v8f wmma_bf16(v16bf a, v16bf b, v8f c) {
  // (neg_a, A, neg_b, B, c_mod, C, reuse_a, reuse_b)
  return __builtin_amdgcn_wmma_f32_16x16x32_bf16(false, a, false, b, (short)0, c, false, false);
}

// ---------------------------------------------------------------------------
// fp32 -> bf16 conversion (x, context)
// ---------------------------------------------------------------------------
__global__ void f32_to_bf16_kernel(const float* __restrict__ in,
                                   unsigned short* __restrict__ out, int n) {
  int i = blockIdx.x * blockDim.x + threadIdx.x;
  int stride = gridDim.x * blockDim.x;
  for (; i < n; i += stride) out[i] = f2bf(in[i]);
}

// ---------------------------------------------------------------------------
// Weight prep: Wq as-is; Wk_eff/Wv_eff realize the cat(k,v)->(h d j) shuffle
// as a row permutation of [Wk;Wv]:
//   Wk_eff[h*64+d] = src row (h*128+2d)   of [Wk;Wv]
//   Wv_eff[h*64+d] = src row (h*128+2d+1) of [Wk;Wv]
// ---------------------------------------------------------------------------
__global__ void prep_weights_kernel(const float* __restrict__ Wq,
                                    const float* __restrict__ Wk,
                                    const float* __restrict__ Wv,
                                    unsigned short* __restrict__ wqb,
                                    unsigned short* __restrict__ wkb,
                                    unsigned short* __restrict__ wvb) {
  int i = blockIdx.x * blockDim.x + threadIdx.x;
  const int total = DIM * DIM;
  int stride = gridDim.x * blockDim.x;
  for (; i < total; i += stride) {
    int j = i / DIM;                 // output channel
    int c = i - j * DIM;             // input channel
    wqb[i] = f2bf(Wq[i]);
    int d = j & (HD - 1), h = j >> 6;
    int src = h * 128 + 2 * d;       // row in cat(Wk, Wv)
    const float* rk = (src < DIM) ? (Wk + (size_t)src * DIM)
                                  : (Wv + (size_t)(src - DIM) * DIM);
    wkb[i] = f2bf(rk[c]);
    int src2 = src + 1;
    const float* rv = (src2 < DIM) ? (Wk + (size_t)src2 * DIM)
                                   : (Wv + (size_t)(src2 - DIM) * DIM);
    wvb[i] = f2bf(rv[c]);
  }
}

// ---------------------------------------------------------------------------
// WMMA GEMM:  Y[row, col] = sum_k A[row,k] * W[col,k]   (both bf16 row-major)
// Block = 128 threads = 4 waves; each wave owns 16 rows x 64 cols (one head).
// k-loop is register double-buffered: chunk k0+32 loads overlap chunk k0's
// WMMAs, so the waits before each WMMA only cover already-landed loads.
// mode 0/1: fused per-head RMSNorm epilogue, bf16 row-major store.
// mode 2  : no norm, bf16 store transposed to (b, col, m) with m contiguous.
// ---------------------------------------------------------------------------
__global__ __launch_bounds__(128)
void gemm_bf16_wmma(const unsigned short* __restrict__ A,
                    const unsigned short* __restrict__ W,
                    unsigned short* __restrict__ Out, int mode) {
  const int tid  = threadIdx.x;
  const int wave = tid >> 5;
  const int lane = tid & 31;
  const int l15  = lane & 15;
  const int rowbase = blockIdx.x * 64 + wave * 16;
  const int colbase = blockIdx.y * 64;

  const unsigned short* Arow = A + (size_t)(rowbase + l15) * DIM;
  const unsigned short* Wb0 = W + (size_t)(colbase + l15)      * DIM;
  const unsigned short* Wb1 = W + (size_t)(colbase + l15 + 16) * DIM;
  const unsigned short* Wb2 = W + (size_t)(colbase + l15 + 32) * DIM;
  const unsigned short* Wb3 = W + (size_t)(colbase + l15 + 48) * DIM;

  v8f acc0 = {}, acc1 = {}, acc2 = {}, acc3 = {};

  // prologue: chunk 0 fragments
  v16bf af = load_fragA(Arow, lane);
  v16bf b0 = load_fragB(Wb0, lane);
  v16bf b1 = load_fragB(Wb1, lane);
  v16bf b2 = load_fragB(Wb2, lane);
  v16bf b3 = load_fragB(Wb3, lane);

  for (int k0 = 32; k0 < DIM; k0 += 32) {
    // issue next chunk's loads first, then WMMA on the current chunk
    v16bf afn = load_fragA(Arow + k0, lane);
    v16bf b0n = load_fragB(Wb0 + k0, lane);
    v16bf b1n = load_fragB(Wb1 + k0, lane);
    v16bf b2n = load_fragB(Wb2 + k0, lane);
    v16bf b3n = load_fragB(Wb3 + k0, lane);
    __builtin_prefetch(Arow + k0 + 96, 0, 1);   // global_prefetch_b8, A stream
    acc0 = wmma_bf16(af, b0, acc0);
    acc1 = wmma_bf16(af, b1, acc1);
    acc2 = wmma_bf16(af, b2, acc2);
    acc3 = wmma_bf16(af, b3, acc3);
    af = afn; b0 = b0n; b1 = b1n; b2 = b2n; b3 = b3n;
  }
  acc0 = wmma_bf16(af, b0, acc0);
  acc1 = wmma_bf16(af, b1, acc1);
  acc2 = wmma_bf16(af, b2, acc2);
  acc3 = wmma_bf16(af, b3, acc3);

  // Per-head RMSNorm: the 64-wide tile is exactly one head.
  float scale[8];
  if (mode != 2) {
#pragma unroll
    for (int r = 0; r < 8; ++r) {
      float s = acc0[r] * acc0[r] + acc1[r] * acc1[r] +
                acc2[r] * acc2[r] + acc3[r] * acc3[r];
      s += __shfl_xor(s, 1, 32);
      s += __shfl_xor(s, 2, 32);
      s += __shfl_xor(s, 4, 32);
      s += __shfl_xor(s, 8, 32);
      scale[r] = rsqrtf(s * (1.0f / HD) + 1e-6f);
    }
  } else {
#pragma unroll
    for (int r = 0; r < 8; ++r) scale[r] = 1.0f;
  }

  const int rhalf = (lane & 16) ? 8 : 0;
#pragma unroll
  for (int r = 0; r < 8; ++r) {
    const int rowg = rowbase + rhalf + r;
    const float v0 = acc0[r] * scale[r];
    const float v1 = acc1[r] * scale[r];
    const float v2 = acc2[r] * scale[r];
    const float v3 = acc3[r] * scale[r];
    const int colg = colbase + l15;
    if (mode == 2) {
      const int bb = rowg >> 9;          // / NM
      const int mm = rowg & (NM - 1);
      unsigned short* ob = Out + (size_t)bb * DIM * NM + mm;
      ob[(size_t)(colg +  0) * NM] = f2bf(v0);
      ob[(size_t)(colg + 16) * NM] = f2bf(v1);
      ob[(size_t)(colg + 32) * NM] = f2bf(v2);
      ob[(size_t)(colg + 48) * NM] = f2bf(v3);
    } else {
      unsigned short* orow = Out + (size_t)rowg * DIM + colg;
      orow[0]  = f2bf(v0);
      orow[16] = f2bf(v1);
      orow[32] = f2bf(v2);
      orow[48] = f2bf(v3);
    }
  }
}

// ---------------------------------------------------------------------------
// Flash attention: 4 waves/block, each wave owns 16 query rows of one (b,h).
// m=512 in 32-wide chunks: 4 WMMA for QK^T, online softmax in registers,
// P staged via per-wave LDS (D-layout -> A-fragment layout), 4 WMMA for P@V.
// Pipelined: V-fragment loads issue at top of the iteration (in flight during
// softmax), next chunk's K fragments issue right after the score WMMAs.
// ---------------------------------------------------------------------------
__global__ __launch_bounds__(128)
void attn_wmma(const unsigned short* __restrict__ Q,
               const unsigned short* __restrict__ K,
               const unsigned short* __restrict__ VT,
               float* __restrict__ Out) {
  __shared__ __align__(64) unsigned short lds_p[4][16][32];

  const int tid  = threadIdx.x;
  const int wave = tid >> 5;
  const int lane = tid & 31;
  const int l15  = lane & 15;
  const int bh = blockIdx.y, b = bh >> 5, h = bh & (NH - 1);
  const int n0 = blockIdx.x * 64 + wave * 16;

  const unsigned short* qrow = Q + (size_t)(b * NQ + n0 + l15) * DIM + h * HD;
  const v16bf qa0 = load_fragA(qrow, lane);        // d 0..31
  const v16bf qa1 = load_fragA(qrow + 32, lane);   // d 32..63

  float mrun[8], lrun[8];
#pragma unroll
  for (int r = 0; r < 8; ++r) { mrun[r] = -3.0e38f; lrun[r] = 0.0f; }
  v8f o0 = {}, o1 = {}, o2 = {}, o3 = {};

  const unsigned short* kbase = K + (size_t)(b * NM) * DIM + h * HD;
  const unsigned short* vbase = VT + (size_t)(b * DIM + h * HD) * NM;
  const float sm_scale = 0.125f;                   // 1/sqrt(64)

  // prologue: K fragments for chunk m0 = 0
  v16bf kf00, kf01, kf10, kf11;
  {
    const unsigned short* kp0 = kbase + (size_t)l15 * DIM;
    const unsigned short* kp1 = kbase + (size_t)(16 + l15) * DIM;
    kf00 = load_fragB(kp0,      lane);
    kf01 = load_fragB(kp0 + 32, lane);
    kf10 = load_fragB(kp1,      lane);
    kf11 = load_fragB(kp1 + 32, lane);
  }

  for (int mc = 0; mc < NM / 32; ++mc) {
    const int m0 = mc * 32;

    // ---- V fragments for this chunk: issue early, consumed after softmax --
    const unsigned short* vp = vbase + m0;
    v16bf vf0 = load_fragB(vp + (size_t)(l15 +  0) * NM, lane);
    v16bf vf1 = load_fragB(vp + (size_t)(l15 + 16) * NM, lane);
    v16bf vf2 = load_fragB(vp + (size_t)(l15 + 32) * NM, lane);
    v16bf vf3 = load_fragB(vp + (size_t)(l15 + 48) * NM, lane);

    // ---- scores: S = (Q @ K^T) * scale, 16 rows x 32 cols ----
    v8f s0 = {}, s1 = {};
    s0 = wmma_bf16(qa0, kf00, s0);
    s0 = wmma_bf16(qa1, kf01, s0);
    s1 = wmma_bf16(qa0, kf10, s1);
    s1 = wmma_bf16(qa1, kf11, s1);

    // ---- prefetch next chunk's K fragments (overlap with softmax) ----
    if (mc + 1 < NM / 32) {
      const unsigned short* kp0n = kbase + (size_t)(m0 + 32 + l15) * DIM;
      const unsigned short* kp1n = kbase + (size_t)(m0 + 48 + l15) * DIM;
      kf00 = load_fragB(kp0n,      lane);
      kf01 = load_fragB(kp0n + 32, lane);
      kf10 = load_fragB(kp1n,      lane);
      kf11 = load_fragB(kp1n + 32, lane);
    }

    // ---- online softmax over the 32-wide chunk (row stats per vgpr r) ----
    const int rowl = ((lane & 16) ? 8 : 0);
#pragma unroll
    for (int r = 0; r < 8; ++r) {
      float a0 = s0[r] * sm_scale;
      float a1 = s1[r] * sm_scale;
      float cm = fmaxf(a0, a1);
      cm = fmaxf(cm, __shfl_xor(cm, 1, 32));
      cm = fmaxf(cm, __shfl_xor(cm, 2, 32));
      cm = fmaxf(cm, __shfl_xor(cm, 4, 32));
      cm = fmaxf(cm, __shfl_xor(cm, 8, 32));
      const float mnew = fmaxf(mrun[r], cm);
      const float corr = __expf(mrun[r] - mnew);
      mrun[r] = mnew;
      const float p0 = __expf(a0 - mnew);
      const float p1 = __expf(a1 - mnew);
      float rs = p0 + p1;
      rs += __shfl_xor(rs, 1, 32);
      rs += __shfl_xor(rs, 2, 32);
      rs += __shfl_xor(rs, 4, 32);
      rs += __shfl_xor(rs, 8, 32);
      lrun[r] = lrun[r] * corr + rs;
      o0[r] *= corr; o1[r] *= corr; o2[r] *= corr; o3[r] *= corr;
      // stage P (D-layout: row = r + 8*half, col = lane&15 + 16*tile)
      lds_p[wave][r + rowl][l15]      = f2bf(p0);
      lds_p[wave][r + rowl][l15 + 16] = f2bf(p1);
    }
    // reload P as an A-fragment (16x32, K = local m). LDS ops from one wave
    // are in-order (DScnt), no barrier needed for intra-wave exchange.
    const v16bf pf = load_fragA(&lds_p[wave][l15][0], lane);

    // ---- O += P @ V : V^T columns (fixed d, m contiguous) are B-fragments --
    o0 = wmma_bf16(pf, vf0, o0);
    o1 = wmma_bf16(pf, vf1, o1);
    o2 = wmma_bf16(pf, vf2, o2);
    o3 = wmma_bf16(pf, vf3, o3);
  }

  // ---- finalize: O /= l, store fp32 (b, n, h*64 + d) ----
  float* obase = Out + (size_t)(b * NQ + n0) * DIM + h * HD + l15;
  const int rowl = ((lane & 16) ? 8 : 0);
#pragma unroll
  for (int r = 0; r < 8; ++r) {
    const float inv = 1.0f / lrun[r];
    float* op = obase + (size_t)(r + rowl) * DIM;
    op[0]  = o0[r] * inv;
    op[16] = o1[r] * inv;
    op[32] = o2[r] * inv;
    op[48] = o3[r] * inv;
  }
}

// ---------------------------------------------------------------------------
// Launch
// ---------------------------------------------------------------------------
extern "C" void kernel_launch(void* const* d_in, const int* in_sizes, int n_in,
                              void* d_out, int out_size, void* d_ws, size_t ws_size,
                              hipStream_t stream) {
  const float* x   = (const float*)d_in[0];
  const float* ctx = (const float*)d_in[1];
  const float* Wq  = (const float*)d_in[2];
  const float* Wk  = (const float*)d_in[3];
  const float* Wv  = (const float*)d_in[4];
  float* out = (float*)d_out;

  char* ws = (char*)d_ws;
  size_t off = 0;
  unsigned short* XB  = (unsigned short*)(ws + off); off += (size_t)NB * NQ * DIM * 2;  // x bf16
  unsigned short* CB  = (unsigned short*)(ws + off); off += (size_t)NB * NM * DIM * 2;  // ctx bf16
  unsigned short* WQB = (unsigned short*)(ws + off); off += (size_t)DIM * DIM * 2;
  unsigned short* WKB = (unsigned short*)(ws + off); off += (size_t)DIM * DIM * 2;      // permuted
  unsigned short* WVB = (unsigned short*)(ws + off); off += (size_t)DIM * DIM * 2;      // permuted
  unsigned short* QB  = (unsigned short*)(ws + off); off += (size_t)NB * NQ * DIM * 2;  // rmsnorm(q)
  unsigned short* KB  = (unsigned short*)(ws + off); off += (size_t)NB * NM * DIM * 2;  // rmsnorm(k)
  unsigned short* VTB = (unsigned short*)(ws + off); off += (size_t)NB * NM * DIM * 2;  // v^T (b,hd,m)
  (void)off; (void)ws_size; (void)in_sizes; (void)n_in; (void)out_size;

  f32_to_bf16_kernel<<<2048, 256, 0, stream>>>(x,   XB, NB * NQ * DIM);
  f32_to_bf16_kernel<<<512,  256, 0, stream>>>(ctx, CB, NB * NM * DIM);
  prep_weights_kernel<<<2048, 256, 0, stream>>>(Wq, Wk, Wv, WQB, WKB, WVB);

  gemm_bf16_wmma<<<dim3(NB * NQ / 64, DIM / 64), 128, 0, stream>>>(XB, WQB, QB, 0);
  gemm_bf16_wmma<<<dim3(NB * NM / 64, DIM / 64), 128, 0, stream>>>(CB, WKB, KB, 1);
  gemm_bf16_wmma<<<dim3(NB * NM / 64, DIM / 64), 128, 0, stream>>>(CB, WVB, VTB, 2);

  attn_wmma<<<dim3(NQ / 64, NB * NH), 128, 0, stream>>>(QB, KB, VTB, out);
}
